// SelfAttention_71159018160681
// MI455X (gfx1250) — matrix operand order repaired and checked by reading, not compiled, not run on hardware
//
#include <hip/hip_runtime.h>
#include <hip/hip_bf16.h>

// Problem constants
#define BB 4
#define LL 2048
#define DD 1024
#define MM (BB * LL)   // 8192 rows for x/Q/K/V/AO

typedef __attribute__((ext_vector_type(16))) __bf16 bf16x16;
typedef __attribute__((ext_vector_type(8)))  __bf16 bf16x8;
typedef __attribute__((ext_vector_type(8)))  float  f32x8;

// ---------------------------------------------------------------------------
// CDNA5 async copy: 16 bytes global -> LDS per lane, tracked by ASYNCcnt.
// ---------------------------------------------------------------------------
__device__ __forceinline__ void async_copy16(unsigned lds_off, const void* gaddr) {
    asm volatile("global_load_async_to_lds_b128 %0, %1, off"
                 :: "v"(lds_off), "v"(gaddr)
                 : "memory");
}
__device__ __forceinline__ void wait_asynccnt0() {
    asm volatile("s_wait_asynccnt 0x0" ::: "memory");
}
// Low 32 bits of a generic pointer to __shared__ = LDS byte address.
__device__ __forceinline__ unsigned lds_addr(const void* p) {
    return (unsigned)(size_t)p;
}

// ---------------------------------------------------------------------------
// Fragment loads from LDS tiles (wave32 v_wmma_f32_16x16x32_bf16 layouts).
// LDS tile rows have an 80B pitch (64B of k-data + 16B pad) -> the 16-lane
// 16B reads below touch all 64 banks exactly once (bank step 20/row).
// A frag: lane l (r=l&15, h=l>>4): elems 0..7 = k h*8+0..7   (bytes 16h..)
//                                  elems 8..15 = k 16+h*8+.. (bytes 32+16h..)
// B frag: lane l (n=l&15, h=l>>4): elems 0..7 = k h*8+0..7   (bytes 16h..)
//                                  elems 8..15 = k 16+h*8+.. (bytes 32+16h..)
// ---------------------------------------------------------------------------
#define LDSROW 80
__device__ __forceinline__ bf16x16 lds_a_frag(const unsigned char* base, int row0) {
    const int lane = threadIdx.x & 31;
    const int r = lane & 15, h = lane >> 4;
    const unsigned char* p = base + (row0 + r) * LDSROW + h * 16;
    bf16x8 lo = *(const bf16x8*)p;
    bf16x8 hi = *(const bf16x8*)(p + 32);
    bf16x16 f;
#pragma unroll
    for (int i = 0; i < 8; ++i) { f[i] = lo[i]; f[8 + i] = hi[i]; }
    return f;
}
__device__ __forceinline__ bf16x16 lds_b_frag(const unsigned char* base, int row0) {
    const int lane = threadIdx.x & 31;
    const int n = lane & 15, h = lane >> 4;
    const unsigned char* p = base + (row0 + n) * LDSROW + h * 16;
    bf16x8 lo = *(const bf16x8*)p;
    bf16x8 hi = *(const bf16x8*)(p + 32);
    bf16x16 f;
#pragma unroll
    for (int i = 0; i < 8; ++i) { f[i] = lo[i]; f[8 + i] = hi[i]; }
    return f;
}

// ---------------------------------------------------------------------------
// Generic bf16 GEMM with f32 accumulation:  C[M,N] = A[M,K] @ Bt[N,K]^T
// MODE 0: f32 out + bias               (final projection)
// MODE 1: bf16 out + bias              (Q, K projections)
// MODE 2: bf16 out + bias, stored transposed per batch -> Vt[B][D][L]
// MODE 3: f32 out * scale, no bias     (attention scores)
// MODE 4: bf16 out, no bias            (P @ V)
//
// Block = 256 threads = 8 waves (4 waveM x 2 waveN); wave tile 64x64
// (4x4 accumulators, 16 WMMA per K-step). Block tile 256x128.
// A(256x32) + B(128x32) bf16 tiles are staged in LDS per K-step via
// global_load_async_to_lds_b128, double-buffered (2 x 30KB LDS).
// M % 256 == 0, N % 128 == 0, K % 32 == 0.
// ---------------------------------------------------------------------------
template <int MODE>
__launch_bounds__(256)
__global__ void gemm_wmma(const __bf16* __restrict__ A,
                          const __bf16* __restrict__ Bt,
                          const float* __restrict__ bias, float scale,
                          float* __restrict__ outF, __bf16* __restrict__ outB,
                          int M, int N, int K) {
    (void)M;
    constexpr int ASZ = 256 * LDSROW;    // 20480 B
    constexpr int BSZ = 128 * LDSROW;    // 10240 B
    __shared__ unsigned char smem[2][ASZ + BSZ];

    const int t      = threadIdx.x;
    const int lane   = t & 31;
    const int wave   = t >> 5;            // 0..7
    const int waveM  = wave >> 1;         // 0..3
    const int waveN  = wave & 1;          // 0..1
    const int blockM = blockIdx.y * 256;
    const int blockN = blockIdx.x * 128;

    // Stage one K-step's A/B tiles into LDS buffer `buf` (whole block
    // cooperates; 16B chunk per async op; 6 ops per thread).
    auto stage = [&](int buf, int kb) {
#pragma unroll
        for (int i = 0; i < 4; ++i) {                // A: 1024 chunks
            const int c   = t + i * 256;
            const int row = c >> 2, cir = c & 3;
            const __bf16* g = A + (size_t)(blockM + row) * K + kb + cir * 8;
            async_copy16(lds_addr(&smem[buf][row * LDSROW + cir * 16]), g);
        }
#pragma unroll
        for (int i = 0; i < 2; ++i) {                // B: 512 chunks
            const int c   = t + i * 256;
            const int row = c >> 2, cir = c & 3;
            const __bf16* g = Bt + (size_t)(blockN + row) * K + kb + cir * 8;
            async_copy16(lds_addr(&smem[buf][ASZ + row * LDSROW + cir * 16]), g);
        }
    };

    f32x8 acc[4][4];
#pragma unroll
    for (int i = 0; i < 4; ++i)
#pragma unroll
        for (int j = 0; j < 4; ++j)
            acc[i][j] = (f32x8){0.f, 0.f, 0.f, 0.f, 0.f, 0.f, 0.f, 0.f};

    stage(0, 0);
    int cur = 0;
    for (int kb = 0; kb < K; kb += 32) {
        wait_asynccnt0();        // my staging of smem[cur] (issued last iter) done
        __syncthreads();         // all waves staged + finished reading smem[cur^1]
        if (kb + 32 < K) stage(cur ^ 1, kb + 32);

        const unsigned char* aBuf = &smem[cur][0];
        const unsigned char* bBuf = &smem[cur][ASZ];
        bf16x16 a[4], b[4];
#pragma unroll
        for (int i = 0; i < 4; ++i) a[i] = lds_a_frag(aBuf, waveM * 64 + 16 * i);
#pragma unroll
        for (int j = 0; j < 4; ++j) b[j] = lds_b_frag(bBuf, waveN * 64 + 16 * j);

#pragma unroll
        for (int i = 0; i < 4; ++i)
#pragma unroll
            for (int j = 0; j < 4; ++j)
                acc[i][j] = __builtin_amdgcn_wmma_f32_16x16x32_bf16(
                    false, a[i], false, b[j], (short)0, acc[i][j], false, false);
        cur ^= 1;
    }

    // C/D layout: VGPR vg, lane (cn=lane&15, h=lane>>4) holds
    // row = vg + 8*h, col = cn within each 16x16 tile.
    const int tileM = blockM + waveM * 64;
    const int tileN = blockN + waveN * 64;
    const int cn = lane & 15;
    const int h  = lane >> 4;
#pragma unroll
    for (int i = 0; i < 4; ++i) {
#pragma unroll
        for (int j = 0; j < 4; ++j) {
            const int n = tileN + j * 16 + cn;
            float bval = 0.f;
            if constexpr (MODE == 0 || MODE == 1 || MODE == 2) bval = bias[n];
#pragma unroll
            for (int vg = 0; vg < 8; ++vg) {
                const int m = tileM + i * 16 + vg + 8 * h;
                const float val = acc[i][j][vg] + bval;
                if constexpr (MODE == 0) {
                    outF[(size_t)m * N + n] = val;
                } else if constexpr (MODE == 1) {
                    outB[(size_t)m * N + n] = (__bf16)val;
                } else if constexpr (MODE == 2) {
                    const int b  = m >> 11;         // m / L
                    const int ml = m & (LL - 1);    // m % L
                    outB[(size_t)b * DD * LL + (size_t)n * LL + ml] = (__bf16)val;
                } else if constexpr (MODE == 3) {
                    outF[(size_t)m * N + n] = val * scale;
                } else {
                    outB[(size_t)m * N + n] = (__bf16)val;
                }
            }
        }
    }
}

// ---------------------------------------------------------------------------
// f32 -> bf16 convert
// ---------------------------------------------------------------------------
__global__ void cvt_f32_bf16(const float* __restrict__ src,
                             __bf16* __restrict__ dst, int n) {
    int i = blockIdx.x * blockDim.x + threadIdx.x;
    if (i < n) dst[i] = (__bf16)src[i];
}

// W[K][N] (f32) -> Wt[N][K] (bf16), K = N = DD
__global__ void cvt_transpose_bf16(const float* __restrict__ W,
                                   __bf16* __restrict__ Wt) {
    int n = blockIdx.x * 16 + threadIdx.x;
    int k = blockIdx.y * 16 + threadIdx.y;
    Wt[(size_t)n * DD + k] = (__bf16)W[(size_t)k * DD + n];
}

// ---------------------------------------------------------------------------
// Row softmax on f32 scores [LL x LL] -> bf16 probabilities.
// One 256-thread block per row; 8 elements per thread held in registers.
// ---------------------------------------------------------------------------
__launch_bounds__(256)
__global__ void softmax_rows(const float* __restrict__ S,
                             __bf16* __restrict__ P) {
    const int row = blockIdx.x;
    const float* s = S + (size_t)row * LL;
    __bf16* p = P + (size_t)row * LL;
    const int t = threadIdx.x;

    float v[8];
    float mx = -3.4e38f;
#pragma unroll
    for (int i = 0; i < 8; ++i) {
        v[i] = s[t + i * 256];
        mx = fmaxf(mx, v[i]);
    }
    __shared__ float red[256];
    red[t] = mx;
    __syncthreads();
    for (int off = 128; off > 0; off >>= 1) {
        if (t < off) red[t] = fmaxf(red[t], red[t + off]);
        __syncthreads();
    }
    mx = red[0];
    __syncthreads();

    float sum = 0.f;
#pragma unroll
    for (int i = 0; i < 8; ++i) {
        v[i] = __expf(v[i] - mx);
        sum += v[i];
    }
    red[t] = sum;
    __syncthreads();
    for (int off = 128; off > 0; off >>= 1) {
        if (t < off) red[t] += red[t + off];
        __syncthreads();
    }
    const float inv = 1.f / red[0];
#pragma unroll
    for (int i = 0; i < 8; ++i) p[t + i * 256] = (__bf16)(v[i] * inv);
}

// ---------------------------------------------------------------------------
// Host-side launch
// ---------------------------------------------------------------------------
extern "C" void kernel_launch(void* const* d_in, const int* in_sizes, int n_in,
                              void* d_out, int out_size, void* d_ws, size_t ws_size,
                              hipStream_t stream) {
    (void)in_sizes; (void)n_in; (void)out_size; (void)ws_size;

    const float* x  = (const float*)d_in[0];
    const float* Wq = (const float*)d_in[1];
    const float* bq = (const float*)d_in[2];
    const float* Wk = (const float*)d_in[3];
    const float* bk = (const float*)d_in[4];
    const float* Wv = (const float*)d_in[5];
    const float* bv = (const float*)d_in[6];
    const float* Wo = (const float*)d_in[7];
    const float* bo = (const float*)d_in[8];
    float* out = (float*)d_out;

    // Workspace layout (bytes)
    constexpr size_t SZ_XB  = (size_t)MM * DD * 2;   // 16 MiB
    constexpr size_t SZ_W   = (size_t)DD * DD * 2;   //  2 MiB each
    constexpr size_t SZ_QKV = (size_t)MM * DD * 2;   // 16 MiB each
    constexpr size_t SZ_S   = (size_t)LL * LL * 4;   // 16 MiB (per-batch reuse)
    constexpr size_t SZ_P   = (size_t)LL * LL * 2;   //  8 MiB (per-batch reuse)

    char* ws = (char*)d_ws;
    size_t off = 0;
    __bf16* xb  = (__bf16*)(ws + off); off += SZ_XB;
    __bf16* wqb = (__bf16*)(ws + off); off += SZ_W;
    __bf16* wkb = (__bf16*)(ws + off); off += SZ_W;
    __bf16* wvb = (__bf16*)(ws + off); off += SZ_W;
    __bf16* wob = (__bf16*)(ws + off); off += SZ_W;
    __bf16* Qb  = (__bf16*)(ws + off); off += SZ_QKV;
    __bf16* Kb  = (__bf16*)(ws + off); off += SZ_QKV;
    __bf16* Vt  = (__bf16*)(ws + off); off += SZ_QKV;   // [B][D][L]
    __bf16* AO  = (__bf16*)(ws + off); off += SZ_QKV;   // [B*L][D]
    float*  S   = (float*)(ws + off);  off += SZ_S;     // [L][L], reused per batch
    __bf16* P   = (__bf16*)(ws + off); off += SZ_P;     // [L][L], reused per batch

    const float scale = 0.03125f;   // 1/sqrt(1024)

    // 1) Convert inputs to bf16 (weights transposed to [N][K])
    {
        const int nX = MM * DD;
        cvt_f32_bf16<<<nX / 256, 256, 0, stream>>>(x, xb, nX);
        dim3 tb(16, 16), tg(DD / 16, DD / 16);
        cvt_transpose_bf16<<<tg, tb, 0, stream>>>(Wq, wqb);
        cvt_transpose_bf16<<<tg, tb, 0, stream>>>(Wk, wkb);
        cvt_transpose_bf16<<<tg, tb, 0, stream>>>(Wv, wvb);
        cvt_transpose_bf16<<<tg, tb, 0, stream>>>(Wo, wob);
    }

    // 2) QKV projections (V stored transposed per batch for the PV GEMM)
    {
        dim3 g(DD / 128, MM / 256);
        gemm_wmma<1><<<g, 256, 0, stream>>>(xb, wqb, bq, 1.f, nullptr, Qb, MM, DD, DD);
        gemm_wmma<1><<<g, 256, 0, stream>>>(xb, wkb, bk, 1.f, nullptr, Kb, MM, DD, DD);
        gemm_wmma<2><<<g, 256, 0, stream>>>(xb, wvb, bv, 1.f, nullptr, Vt, MM, DD, DD);
    }

    // 3) Per-batch attention: scores -> softmax -> P@V
    for (int b = 0; b < BB; ++b) {
        const __bf16* Qbb = Qb + (size_t)b * LL * DD;
        const __bf16* Kbb = Kb + (size_t)b * LL * DD;   // [L][D] == Bt layout for Q@K^T
        const __bf16* Vtb = Vt + (size_t)b * DD * LL;   // [D][L] == Bt layout for P@V
        __bf16* AOb = AO + (size_t)b * LL * DD;

        gemm_wmma<3><<<dim3(LL / 128, LL / 256), 256, 0, stream>>>(
            Qbb, Kbb, nullptr, scale, S, nullptr, LL, LL, DD);
        softmax_rows<<<LL, 256, 0, stream>>>(S, P);
        gemm_wmma<4><<<dim3(DD / 128, LL / 256), 256, 0, stream>>>(
            P, Vtb, nullptr, 1.f, nullptr, AOb, LL, DD, LL);
    }

    // 4) Output projection in f32
    gemm_wmma<0><<<dim3(DD / 128, MM / 256), 256, 0, stream>>>(
        AO, wob, bo, 1.f, out, nullptr, MM, DD, DD);
}
